// PatchedSelfAttention_59897613910440
// MI455X (gfx1250) — matrix operand order repaired and checked by reading, not compiled, not run on hardware
//
#include <hip/hip_runtime.h>
#include <hip/hip_bf16.h>

typedef __bf16 bf16;
typedef __attribute__((ext_vector_type(16))) __bf16 v16bf;
typedef __attribute__((ext_vector_type(8)))  float  v8f;
typedef __attribute__((ext_vector_type(4)))  int    v4i;
typedef __attribute__((address_space(1))) v4i g_v4i;   // global int4
typedef __attribute__((address_space(3))) v4i l_v4i;   // LDS int4

#define NUM_HEADS 32
#define KV_HEADS  2
#define HEAD_DIM  128
#define HIDDEN    4096
#define NUM_SEQS  8
#define SEQ       1024
#define ROT       (HEAD_DIM / 2)                              // 64
#define QKV_OUT   ((NUM_HEADS + 2 * KV_HEADS) * HEAD_DIM)     // 4608
#define SQ        (NUM_SEQS * SEQ)                            // 8192

#if defined(__AMDGCN__) && __has_builtin(__builtin_amdgcn_global_load_async_to_lds_b128)
#define USE_ASYNC_COPY 1
#else
#define USE_ASYNC_COPY 0
#endif

__device__ __forceinline__ v8f wmma_bf16(v16bf a, v16bf b, v8f c) {
  // D(f32 16x16) = A(bf16 16x32) * B(bf16 32x16) + C
  return __builtin_amdgcn_wmma_f32_16x16x32_bf16(false, a, false, b, (short)0, c,
                                                 false, false);
}

__device__ __forceinline__ unsigned pack2(bf16 lo, bf16 hi) {
  union { bf16 h[2]; unsigned u; } x;
  x.h[0] = lo; x.h[1] = hi;
  return x.u;
}

// 16 source elements -> 16 bf16 in a 32-byte register block
__device__ __forceinline__ void load16_bf16(const float* p, bf16* out) {
  const float4* p4 = (const float4*)p;
#pragma unroll
  for (int i = 0; i < 4; ++i) {
    float4 x = p4[i];
    out[4 * i + 0] = (bf16)x.x; out[4 * i + 1] = (bf16)x.y;
    out[4 * i + 2] = (bf16)x.z; out[4 * i + 3] = (bf16)x.w;
  }
}
__device__ __forceinline__ void load16_bf16(const bf16* p, bf16* out) {
  ((uint4*)out)[0] = ((const uint4*)p)[0];
  ((uint4*)out)[1] = ((const uint4*)p)[1];
}

// ---------------------------------------------------------------------------
// Tiled WMMA GEMM: C[M,N] = A[M,K] * B[K,N] (+bias, scatter epilogue for QKV)
// 256 threads = 8 waves; tile 128x128, K-step 32; wave owns 64x32 (4x2 WMMA).
// ---------------------------------------------------------------------------
template <bool QKV_EPI, typename TA>
__global__ __launch_bounds__(256) void gemm_wmma(
    const TA* __restrict__ A, const float* __restrict__ B,
    const float* __restrict__ bias, float* __restrict__ Cout,
    bf16* __restrict__ qb, bf16* __restrict__ kbf, bf16* __restrict__ vbf,
    int M, int N, int K) {
  __shared__ bf16 As[128][32];   // [m][k]
  __shared__ bf16 Bs[128][32];   // transposed [n][k] -> contiguous B frags

  const int tid  = threadIdx.x;
  const int lane = tid & 31;
  const int wave = tid >> 5;
  const int wm   = wave >> 2;    // 0..1
  const int wn   = wave & 3;     // 0..3
  const int m0   = blockIdx.x * 128;
  const int n0   = blockIdx.y * 128;

  const int fm   = lane & 15;
  const int kb8  = (lane >> 4) * 8;
  const int kb16 = (lane >> 4) * 16;

  v8f acc[4][2];
#pragma unroll
  for (int i = 0; i < 4; ++i)
#pragma unroll
    for (int j = 0; j < 2; ++j)
#pragma unroll
      for (int r = 0; r < 8; ++r) acc[i][j][r] = 0.0f;

  const int lrow = tid >> 1;            // A stage: row 0..127
  const int lcol = (tid & 1) * 16;      //          k-offset {0,16}
  const int bk2  = (tid >> 4) * 2;      // B stage: even k-row 0..30
  const int bn8  = (tid & 15) * 8;      //          n-offset 0..120

  for (int k0 = 0; k0 < K; k0 += 32) {
    if (k0 + 32 < K) {                  // gfx1250 global_prefetch_b8
      __builtin_prefetch(&A[(size_t)(m0 + lrow) * K + k0 + 32 + lcol], 0, 1);
      __builtin_prefetch(&B[(size_t)(k0 + 32 + bk2) * N + n0 + bn8], 0, 1);
    }
    {  // A tile: 16 elems/thread, vector load + 2x ds_store_b128
      bf16 abuf[16];
      load16_bf16(&A[(size_t)(m0 + lrow) * K + k0 + lcol], abuf);
      uint4* d = (uint4*)&As[lrow][lcol];
      d[0] = ((uint4*)abuf)[0];
      d[1] = ((uint4*)abuf)[1];
    }
    {  // B tile transposed: pack (k,k+1) pairs -> 8x ds_store_b32
      const float* b0p = B + (size_t)(k0 + bk2) * N + n0 + bn8;
      const float* b1p = b0p + N;
      float4 a0 = ((const float4*)b0p)[0], a1 = ((const float4*)b0p)[1];
      float4 c0 = ((const float4*)b1p)[0], c1 = ((const float4*)b1p)[1];
      float r0[8] = {a0.x, a0.y, a0.z, a0.w, a1.x, a1.y, a1.z, a1.w};
      float r1[8] = {c0.x, c0.y, c0.z, c0.w, c1.x, c1.y, c1.z, c1.w};
#pragma unroll
      for (int i = 0; i < 8; ++i)
        *(unsigned*)&Bs[bn8 + i][bk2] = pack2((bf16)r0[i], (bf16)r1[i]);
    }
    __syncthreads();

    v16bf af[4];
#pragma unroll
    for (int ms = 0; ms < 4; ++ms) {
      const int row = wm * 64 + ms * 16 + fm;
#pragma unroll
      for (int i = 0; i < 16; ++i) {    // A frag, contiguous k order (same as B): K=kb16+0..15
        af[ms][i] = As[row][kb16 + i];
      }
    }
#pragma unroll
    for (int ns = 0; ns < 2; ++ns) {
      v16bf bfrag;
      const int n = wn * 32 + ns * 16 + fm;
#pragma unroll
      for (int t = 0; t < 16; ++t) bfrag[t] = Bs[n][kb16 + t];
#pragma unroll
      for (int ms = 0; ms < 4; ++ms)
        acc[ms][ns] = wmma_bf16(af[ms], bfrag, acc[ms][ns]);
    }
    __syncthreads();
  }

  const int moff = (lane >> 4) * 8;     // C/D: lanes 16..31 hold M=8..15
  if constexpr (QKV_EPI) {
    // n-tile of 128 == exactly one head (HEAD_DIM==128): restage through LDS,
    // then contiguous 16B stores into the [b][h][s][d] bf16 buffer.
    __shared__ bf16 Cs[128][128];
    const float bias0 = bias[n0 + wn * 32 + fm];
    const float bias1 = bias[n0 + wn * 32 + 16 + fm];
#pragma unroll
    for (int ms = 0; ms < 4; ++ms)
#pragma unroll
      for (int ns = 0; ns < 2; ++ns) {
        const float bb = ns ? bias1 : bias0;
#pragma unroll
        for (int r = 0; r < 8; ++r)
          Cs[wm * 64 + ms * 16 + moff + r][wn * 32 + ns * 16 + fm] =
              (bf16)(acc[ms][ns][r] + bb);
      }
    __syncthreads();
    const int row = tid >> 1, half = (tid & 1) * 64;
    const int gm = m0 + row, b = gm / SEQ, s = gm % SEQ;
    bf16* dstbase;
    if (n0 < NUM_HEADS * HEAD_DIM)
      dstbase = qb + (((size_t)(b * NUM_HEADS + n0 / HEAD_DIM)) * SEQ + s) * HEAD_DIM;
    else if (n0 < (NUM_HEADS + KV_HEADS) * HEAD_DIM)
      dstbase = kbf + (((size_t)(b * KV_HEADS + (n0 - NUM_HEADS * HEAD_DIM) / HEAD_DIM)) * SEQ + s) * HEAD_DIM;
    else
      dstbase = vbf + (((size_t)(b * KV_HEADS + (n0 - (NUM_HEADS + KV_HEADS) * HEAD_DIM) / HEAD_DIM)) * SEQ + s) * HEAD_DIM;
    uint4* d4 = (uint4*)(dstbase + half);
    const uint4* s4 = (const uint4*)&Cs[row][half];
#pragma unroll
    for (int i = 0; i < 8; ++i) d4[i] = s4[i];
  } else {
#pragma unroll
    for (int ms = 0; ms < 4; ++ms)
#pragma unroll
      for (int ns = 0; ns < 2; ++ns)
#pragma unroll
        for (int r = 0; r < 8; ++r) {
          const int gm = m0 + wm * 64 + ms * 16 + moff + r;
          const int gn = n0 + wn * 32 + ns * 16 + fm;
          Cout[(size_t)gm * N + gn] = acc[ms][ns][r];
        }
  }
}

// ---------------------------------------------------------------------------
// RoPE in place on q/k buffers. One thread per (token, head, pair).
// rope_cache: (SQ, 1, ROT/2, 2) fp32 -> flat t*ROT + 2*i {+1}.
// ---------------------------------------------------------------------------
__global__ __launch_bounds__(256) void rope_kernel(
    bf16* __restrict__ qb, bf16* __restrict__ kbf,
    const float* __restrict__ rope_cache) {
  const int PAIRS = ROT / 2;                   // 32
  const int NH    = NUM_HEADS + KV_HEADS;      // 34
  int idx = blockIdx.x * blockDim.x + threadIdx.x;
  if (idx >= SQ * NH * PAIRS) return;
  const int i  = idx % PAIRS;
  const int hh = (idx / PAIRS) % NH;
  const int t  = idx / (PAIRS * NH);
  const int b = t / SEQ, s = t % SEQ;
  const float c  = rope_cache[(size_t)t * ROT + 2 * i];
  const float sn = rope_cache[(size_t)t * ROT + 2 * i + 1];
  bf16* base =
      (hh < NUM_HEADS)
          ? qb + (((size_t)(b * NUM_HEADS + hh)) * SEQ + s) * HEAD_DIM
          : kbf + (((size_t)(b * KV_HEADS + (hh - NUM_HEADS))) * SEQ + s) * HEAD_DIM;
  const float x0 = (float)base[2 * i];
  const float x1 = (float)base[2 * i + 1];
  base[2 * i]     = (bf16)(x0 * c - x1 * sn);
  base[2 * i + 1] = (bf16)(x1 * c + x0 * sn);
}

// ---------------------------------------------------------------------------
// Flash attention, causal, GQA (32 q heads -> 2 kv heads).
// WG = (q-tile 64, head, seq); 128 threads = 4 waves, 16 q-rows/wave.
// ---------------------------------------------------------------------------
__global__ __launch_bounds__(128) void flash_attn(
    const bf16* __restrict__ qbuf, const bf16* __restrict__ kbuf,
    const bf16* __restrict__ vbuf, bf16* __restrict__ ctx) {
  __shared__ bf16 Ks[64 * 128];       // [kv][d]  -> contiguous QK^T B frags
  __shared__ bf16 Vt[128 * 64];       // [d][kv]  -> contiguous P@V  B frags
  __shared__ bf16 Ps[4][16][64];      // per-wave P restage (C -> A layout)

  const int tid = threadIdx.x, lane = tid & 31, wave = tid >> 5;
  const int q0 = blockIdx.x * 64;
  const int h = blockIdx.y, b = blockIdx.z;
  const int kvh = h / (NUM_HEADS / KV_HEADS);
  const int fm = lane & 15, moff = (lane >> 4) * 8;
  const int kb8 = (lane >> 4) * 8, kb16 = (lane >> 4) * 16;
  const float scale = 0.08838834764831845f;  // 1/sqrt(128)

  const bf16* qp =
      qbuf + (((size_t)(b * NUM_HEADS + h)) * SEQ + q0 + wave * 16) * HEAD_DIM;
  v16bf aq[4];
#pragma unroll
  for (int c = 0; c < 4; ++c)
#pragma unroll
    for (int i = 0; i < 16; ++i) {
      aq[c][i] = qp[fm * HEAD_DIM + c * 32 + kb16 + i];
    }

  v8f o[8];
#pragma unroll
  for (int ds = 0; ds < 8; ++ds)
#pragma unroll
    for (int r = 0; r < 8; ++r) o[ds][r] = 0.0f;
  float mr[8], lr[8];
#pragma unroll
  for (int r = 0; r < 8; ++r) { mr[r] = -3.0e38f; lr[r] = 0.0f; }

  const bf16* kp = kbuf + ((size_t)(b * KV_HEADS + kvh)) * SEQ * HEAD_DIM;
  const bf16* vp = vbuf + ((size_t)(b * KV_HEADS + kvh)) * SEQ * HEAD_DIM;

  const int nkt = q0 / 64 + 1;   // causal: tiles up to the diagonal only
  for (int kt = 0; kt < nkt; ++kt) {
    __syncthreads();
    {  // K tile: contiguous 16 KB global->LDS, async path on CDNA5
      const uint4* src = (const uint4*)(kp + (size_t)kt * 64 * HEAD_DIM);
      uint4* dst = (uint4*)Ks;
#if USE_ASYNC_COPY
#pragma unroll
      for (int i = 0; i < 8; ++i) {
        g_v4i* g = (g_v4i*)(unsigned long long)(src + tid + i * 128);
        l_v4i* l = (l_v4i*)(unsigned long long)(dst + tid + i * 128);
        __builtin_amdgcn_global_load_async_to_lds_b128(g, l, 0, 0);
      }
#else
#pragma unroll
      for (int i = 0; i < 8; ++i) dst[tid + i * 128] = src[tid + i * 128];
#endif
    }
    {  // V tile transposed into LDS, packed bf16x2 -> ds_store_b32
      const int rp = tid >> 2;           // kv pair 0..31
      const int dh = (tid & 3) * 32;     // d offset
      const bf16* s0 = vp + (size_t)kt * 64 * HEAD_DIM + (2 * rp) * HEAD_DIM + dh;
      const bf16* s1 = s0 + HEAD_DIM;
      union { bf16 h[32]; uint4 q[4]; } r0, r1;
#pragma unroll
      for (int i = 0; i < 4; ++i) {
        r0.q[i] = ((const uint4*)s0)[i];
        r1.q[i] = ((const uint4*)s1)[i];
      }
#pragma unroll
      for (int d = 0; d < 32; ++d)
        *(unsigned*)&Vt[(dh + d) * 64 + 2 * rp] = pack2(r0.h[d], r1.h[d]);
    }
    if (kt + 1 < nkt) {
      __builtin_prefetch(kp + (size_t)(kt + 1) * 64 * HEAD_DIM + tid * 64, 0, 1);
      __builtin_prefetch(vp + (size_t)(kt + 1) * 64 * HEAD_DIM + tid * 64, 0, 1);
    }
#if USE_ASYNC_COPY
    asm volatile("s_wait_asynccnt 0" ::: "memory");
#endif
    __syncthreads();

    // S = Q K^T (16x64 per wave as 4 n-subtiles), scale + causal mask
    v8f s[4];
#pragma unroll
    for (int j = 0; j < 4; ++j) {
#pragma unroll
      for (int r = 0; r < 8; ++r) s[j][r] = 0.0f;
#pragma unroll
      for (int c = 0; c < 4; ++c) {
        v16bf bk;
#pragma unroll
        for (int t = 0; t < 16; ++t)
          bk[t] = Ks[(j * 16 + fm) * 128 + c * 32 + kb16 + t];
        s[j] = wmma_bf16(aq[c], bk, s[j]);
      }
#pragma unroll
      for (int r = 0; r < 8; ++r) {
        const int qg = q0 + wave * 16 + moff + r;
        const int kg = kt * 64 + j * 16 + fm;
        const float v = s[j][r] * scale;
        s[j][r] = (kg > qg) ? -1.0e30f : v;
      }
    }

    // online softmax: row stats in 16-lane groups (C-layout rows)
    float alpha[8], rsum[8];
#pragma unroll
    for (int r = 0; r < 8; ++r) {
      float v = fmaxf(fmaxf(s[0][r], s[1][r]), fmaxf(s[2][r], s[3][r]));
#pragma unroll
      for (int x = 1; x < 16; x <<= 1) v = fmaxf(v, __shfl_xor(v, x, 32));
      const float mn = fmaxf(mr[r], v);
      alpha[r] = __expf(mr[r] - mn);
      mr[r] = mn;
      rsum[r] = 0.0f;
    }
#pragma unroll
    for (int j = 0; j < 4; ++j)
#pragma unroll
      for (int r = 0; r < 8; ++r) {
        const float p = __expf(s[j][r] - mr[r]);
        s[j][r] = p;
        rsum[r] += p;
      }
#pragma unroll
    for (int r = 0; r < 8; ++r) {
      float v = rsum[r];
#pragma unroll
      for (int x = 1; x < 16; x <<= 1) v += __shfl_xor(v, x, 32);
      lr[r] = lr[r] * alpha[r] + v;
    }
#pragma unroll
    for (int ds = 0; ds < 8; ++ds)
#pragma unroll
      for (int r = 0; r < 8; ++r) o[ds][r] *= alpha[r];

    // restage P (C layout) -> LDS -> A layout; same-wave LDS ops are in order
#pragma unroll
    for (int j = 0; j < 4; ++j)
#pragma unroll
      for (int r = 0; r < 8; ++r)
        Ps[wave][moff + r][j * 16 + fm] = (bf16)s[j][r];

    // O += P V
#pragma unroll
    for (int c = 0; c < 2; ++c) {
      v16bf pa;
#pragma unroll
      for (int i = 0; i < 16; ++i) {
        pa[i] = Ps[wave][fm][c * 32 + kb16 + i];
      }
#pragma unroll
      for (int ds = 0; ds < 8; ++ds) {
        v16bf bv;
#pragma unroll
        for (int t = 0; t < 16; ++t)
          bv[t] = Vt[(ds * 16 + fm) * 64 + c * 32 + kb16 + t];
        o[ds] = wmma_bf16(pa, bv, o[ds]);
      }
    }
  }

  // epilogue: restage O through LDS (reuse Ks) -> contiguous 16B ctx stores
  __syncthreads();
  float inv[8];
#pragma unroll
  for (int r = 0; r < 8; ++r) inv[r] = 1.0f / lr[r];
#pragma unroll
  for (int ds = 0; ds < 8; ++ds)
#pragma unroll
    for (int r = 0; r < 8; ++r)
      Ks[(wave * 16 + moff + r) * 128 + ds * 16 + fm] = (bf16)(o[ds][r] * inv[r]);
  __syncthreads();
  {
    const int row = tid >> 1, half = (tid & 1) * 64;
    const int qg = q0 + row;
    uint4* d4 = (uint4*)(ctx + ((size_t)(b * SEQ + qg)) * HIDDEN + h * HEAD_DIM + half);
    const uint4* s4 = (const uint4*)(Ks + row * 128 + half);
#pragma unroll
    for (int i = 0; i < 8; ++i) d4[i] = s4[i];
  }
}

// ---------------------------------------------------------------------------
extern "C" void kernel_launch(void* const* d_in, const int* in_sizes, int n_in,
                              void* d_out, int out_size, void* d_ws,
                              size_t ws_size, hipStream_t stream) {
  const float* hidden     = (const float*)d_in[0];
  const float* rope_cache = (const float*)d_in[1];
  const float* w_qkv      = (const float*)d_in[2];
  const float* b_qkv      = (const float*)d_in[3];
  const float* w_dense    = (const float*)d_in[4];
  // d_in[5..9]: cache_k/cache_v/block_offsets/seq_lengths — the cache is a
  // pure pass-through in the reference (scatter then gather of identical
  // blocks; only `out` is returned), so K/V stay in workspace.
  float* out = (float*)d_out;

  char* ws = (char*)d_ws;
  size_t off = 0;
  auto alloc = [&](size_t bytes) {
    void* p = ws + off;
    off = (off + bytes + 255) & ~(size_t)255;
    return p;
  };
  bf16* qbuf = (bf16*)alloc((size_t)SQ * NUM_HEADS * HEAD_DIM * sizeof(bf16));
  bf16* kbuf = (bf16*)alloc((size_t)SQ * KV_HEADS * HEAD_DIM * sizeof(bf16));
  bf16* vbuf = (bf16*)alloc((size_t)SQ * KV_HEADS * HEAD_DIM * sizeof(bf16));
  bf16* ctx  = (bf16*)alloc((size_t)SQ * HIDDEN * sizeof(bf16));

  // 1) QKV projection + bias, scattered into [b][h][s][d] bf16 buffers
  dim3 g1(SQ / 128, QKV_OUT / 128);
  gemm_wmma<true, float><<<g1, 256, 0, stream>>>(
      hidden, w_qkv, b_qkv, nullptr, qbuf, kbuf, vbuf, SQ, QKV_OUT, HIDDEN);

  // 2) RoPE in place on q and k
  const int nrope = SQ * (NUM_HEADS + KV_HEADS) * (ROT / 2);
  rope_kernel<<<(nrope + 255) / 256, 256, 0, stream>>>(qbuf, kbuf, rope_cache);

  // 3) causal GQA flash attention
  dim3 g3(SEQ / 64, NUM_HEADS, NUM_SEQS);
  flash_attn<<<g3, 128, 0, stream>>>(qbuf, kbuf, vbuf, ctx);

  // 4) output projection
  dim3 g4(SQ / 128, HIDDEN / 128);
  gemm_wmma<false, bf16><<<g4, 256, 0, stream>>>(
      ctx, w_dense, nullptr, out, nullptr, nullptr, nullptr, SQ, HIDDEN, HIDDEN);
}
